// RelativeMultiHeadAttention_49581102465357
// MI455X (gfx1250) — matrix operand order repaired and checked by reading, not compiled, not run on hardware
//
#include <hip/hip_runtime.h>

#define B_ 4
#define T_ 512
#define TAU_ 512
#define L_ 1024
#define DM_ 1024
#define H_ 16
#define D_ 64

typedef __bf16 bf16;
typedef __attribute__((ext_vector_type(16))) __bf16 v16bf;
typedef __attribute__((ext_vector_type(8)))  __bf16 v8bf;
typedef __attribute__((ext_vector_type(8)))  float  v8f;
typedef __attribute__((ext_vector_type(4)))  unsigned int u32x4;
typedef __attribute__((ext_vector_type(8)))  int i32x8;
typedef __attribute__((ext_vector_type(4)))  int i32x4;

#if __has_builtin(__builtin_amdgcn_tensor_load_to_lds) && __has_builtin(__builtin_amdgcn_s_wait_tensorcnt)
#define TDM_OK 1
#else
#define TDM_OK 0
#endif

// ---------- helpers ----------
__device__ __forceinline__ bf16 f2bf(float f) {
    unsigned u = __builtin_bit_cast(unsigned, f);
    unsigned r = u + 0x7FFFu + ((u >> 16) & 1u);        // round-to-nearest-even
    return __builtin_bit_cast(bf16, (unsigned short)(r >> 16));
}
__device__ __forceinline__ float bf2f(bf16 h) {
    unsigned u = ((unsigned)__builtin_bit_cast(unsigned short, h)) << 16;
    return __builtin_bit_cast(float, u);
}
__device__ __forceinline__ float bperm(float v, int srcLane) {
    int r = __builtin_amdgcn_ds_bpermute(srcLane << 2, __builtin_bit_cast(int, v));
    return __builtin_bit_cast(float, r);
}
__device__ __forceinline__ v8f wmma_bf16(v16bf a, v16bf b, v8f c) {
    return __builtin_amdgcn_wmma_f32_16x16x32_bf16(false, a, false, b, (short)0, c, false, false);
}

#if TDM_OK
// Issue one TDM load of a (rows x 32) bf16 tile (row stride = stride_elems) into LDS,
// packed row-major 64B/row. Descriptor per CDNA5 ISA section 8 (D# groups 0..3).
// clang-23 toolchain: 6-arg builtin (g0, g1, g2, g3, extra_g, cpol).
__device__ __forceinline__ void tdm_load_tile(unsigned int lds_byte, const bf16* gsrc,
                                              int rows, int stride_elems) {
    unsigned long long ga = (unsigned long long)(uintptr_t)gsrc;
    u32x4 g0;
    g0[0] = 1u;                                          // count=1, user descriptor
    g0[1] = lds_byte;                                    // lds_addr
    g0[2] = (unsigned int)ga;                            // global_addr[31:0]
    g0[3] = (unsigned int)((ga >> 32) & 0x1FFFFFFu)      // global_addr[56:32]
          | (2u << 30);                                  // type = 2 ("image")
    unsigned int td0 = (unsigned int)stride_elems;       // tensor_dim0 (>= tile_dim0)
    unsigned int td1 = 0x10000u;                         // tensor_dim1 (generous)
    unsigned int s0  = (unsigned int)stride_elems;       // tensor_dim0_stride
    i32x8 g1;
    g1[0] = (int)(1u << 16);                             // data_size = 1 -> 2 bytes
    g1[1] = (int)((td0 & 0xFFFFu) << 16);                // tensor_dim0[15:0]
    g1[2] = (int)(((td0 >> 16) & 0xFFFFu) | ((td1 & 0xFFFFu) << 16));
    g1[3] = (int)(((td1 >> 16) & 0xFFFFu) | (32u << 16));// tile_dim0 = 32
    g1[4] = (int)((unsigned int)rows & 0xFFFFu);         // tile_dim1 = rows, tile_dim2 = 0
    g1[5] = (int)s0;                                     // stride[31:0]
    g1[6] = 0;                                           // stride[47:32], dim1_stride lo
    g1[7] = 0;
    i32x4 gz4 = {0, 0, 0, 0};
    i32x8 gz8 = {0, 0, 0, 0, 0, 0, 0, 0};
    __builtin_amdgcn_tensor_load_to_lds(g0, g1, gz4, gz4, gz8, 0);
}
#endif

// ---------- LayerNorm over concat(memory, inputs), write bf16 ----------
__global__ void ln_kernel(const float* __restrict__ inputs, const float* __restrict__ memory,
                          const float* __restrict__ gamma, const float* __restrict__ beta,
                          bf16* __restrict__ xln) {
    __shared__ float red[256];
    int row = blockIdx.x;            // b*L + p
    int b = row >> 10, p = row & 1023;
    const float* src = (p < TAU_) ? (memory + ((size_t)b * TAU_ + p) * DM_)
                                  : (inputs + ((size_t)b * T_ + (p - TAU_)) * DM_);
    int tid = threadIdx.x;
    float s = 0.f;
    for (int c = tid; c < DM_; c += 256) s += src[c];
    red[tid] = s; __syncthreads();
    for (int o = 128; o > 0; o >>= 1) { if (tid < o) red[tid] += red[tid + o]; __syncthreads(); }
    float mu = red[0] * (1.0f / DM_);
    __syncthreads();
    float v = 0.f;
    for (int c = tid; c < DM_; c += 256) { float d = src[c] - mu; v += d * d; }
    red[tid] = v; __syncthreads();
    for (int o = 128; o > 0; o >>= 1) { if (tid < o) red[tid] += red[tid + o]; __syncthreads(); }
    float rstd = rsqrtf(red[0] * (1.0f / DM_) + 1e-5f);
    for (int c = tid; c < DM_; c += 256)
        xln[(size_t)row * DM_ + c] = f2bf((src[c] - mu) * rstd * gamma[c] + beta[c]);
}

// ---------- phi = [sin(pos*invf), cos(pos*invf)] ----------
__global__ void phi_kernel(bf16* __restrict__ phi) {
    int p = blockIdx.x, tid = threadIdx.x;
    float pos = (float)(L_ - 1 - p);
    for (int c = tid; c < DM_; c += 256) {
        int cp = (c < 512) ? c : (c - 512);
        float invf = __expf(-(2.0f * (float)cp / 1024.0f) * 9.21034037197618f); // ln(1e4)
        float ang = pos * invf;
        phi[(size_t)p * DM_ + c] = f2bf((c < 512) ? sinf(ang) : cosf(ang));
    }
}

// ---------- f32 -> bf16 convert ----------
__global__ void cvt_kernel(const float* __restrict__ src, bf16* __restrict__ dst, int n) {
    int i = blockIdx.x * 256 + threadIdx.x;
    if (i < n) dst[i] = f2bf(src[i]);
}

// ---------- tiled WMMA GEMM: C(MxN) = A(MxK,row) * B(NxK,row)^T ----------
// Workgroup = 4 waves, 64x64 output tile; K staged 32 at a time through LDS.
// TDM (tensor_load_to_lds) double-buffered when available; sync staging otherwise.
__global__ void __launch_bounds__(128)
gemm_nt(const bf16* __restrict__ A, const bf16* __restrict__ Bm,
        void* __restrict__ C, int M, int N, int K,
        int lda, int ldb, int ldc, int out_f32) {
    __shared__ __align__(64) bf16 lA[2][64][32];
    __shared__ __align__(64) bf16 lB[2][64][32];
    const int tid  = threadIdx.x;
    const int lane = tid & 31;
    const int wave = tid >> 5;
    const int m15 = lane & 15, half = lane >> 4;
    const int kb_a = half * 16, kb_b = half * 16;
    const int tm0 = blockIdx.y * 64;
    const int tn0 = blockIdx.x * 64;
    (void)M; (void)N;

    v8f acc[4] = {{}, {}, {}, {}};
    const int nst = K / 32;
    int buf = 0;

#if TDM_OK
    unsigned int aoff = (unsigned int)(uintptr_t)(void*)&lA[0][0][0];
    unsigned int boff = (unsigned int)(uintptr_t)(void*)&lB[0][0][0];
    if (wave == 0) {
        tdm_load_tile(aoff, A + (size_t)tm0 * lda, 64, lda);
        tdm_load_tile(boff, Bm + (size_t)tn0 * ldb, 64, ldb);
    }
    for (int ks = 0; ks < nst; ++ks) {
        if (wave == 0) {
            if (ks + 1 < nst) {
                int k0 = (ks + 1) * 32;
                tdm_load_tile(aoff + (unsigned)(buf ^ 1) * 4096u,
                              A + (size_t)tm0 * lda + k0, 64, lda);
                tdm_load_tile(boff + (unsigned)(buf ^ 1) * 4096u,
                              Bm + (size_t)tn0 * ldb + k0, 64, ldb);
                __builtin_amdgcn_s_wait_tensorcnt(2);   // stage ks landed
            } else {
                __builtin_amdgcn_s_wait_tensorcnt(0);
            }
        }
        __syncthreads();
        v8bf alo = *(const v8bf*)&lA[buf][wave * 16 + m15][kb_a];
        v8bf ahi = *(const v8bf*)&lA[buf][wave * 16 + m15][kb_a + 8];
        v16bf av;
#pragma unroll
        for (int i = 0; i < 8; ++i) { av[i] = alo[i]; av[i + 8] = ahi[i]; }
#pragma unroll
        for (int nc = 0; nc < 4; ++nc) {
            v16bf bv = *(const v16bf*)&lB[buf][nc * 16 + m15][kb_b];
            acc[nc] = wmma_bf16(av, bv, acc[nc]);
        }
        __syncthreads();
        buf ^= 1;
    }
#else
    for (int ks = 0; ks < nst; ++ks) {
        int k0 = ks * 32;
#pragma unroll
        for (int q = 0; q < 2; ++q) {
            int c = tid + q * 128;                  // 256 chunks of 16B each
            int row = c >> 2, col8 = (c & 3) * 8;
            *(v8bf*)&lA[0][row][col8] = *(const v8bf*)(A + (size_t)(tm0 + row) * lda + k0 + col8);
            *(v8bf*)&lB[0][row][col8] = *(const v8bf*)(Bm + (size_t)(tn0 + row) * ldb + k0 + col8);
        }
        __syncthreads();
        v8bf alo = *(const v8bf*)&lA[0][wave * 16 + m15][kb_a];
        v8bf ahi = *(const v8bf*)&lA[0][wave * 16 + m15][kb_a + 8];
        v16bf av;
#pragma unroll
        for (int i = 0; i < 8; ++i) { av[i] = alo[i]; av[i + 8] = ahi[i]; }
#pragma unroll
        for (int nc = 0; nc < 4; ++nc) {
            v16bf bv = *(const v16bf*)&lB[0][nc * 16 + m15][kb_b];
            acc[nc] = wmma_bf16(av, bv, acc[nc]);
        }
        __syncthreads();
    }
#endif

#pragma unroll
    for (int nc = 0; nc < 4; ++nc)
#pragma unroll
        for (int r = 0; r < 8; ++r) {
            size_t row = (size_t)tm0 + wave * 16 + r + half * 8;
            size_t col = (size_t)tn0 + nc * 16 + m15;
            if (out_f32) ((float*)C)[row * ldc + col] = acc[nc][r];
            else         ((bf16*)C)[row * ldc + col] = f2bf(acc[nc][r]);
        }
}

// ---------- q + u / q + v (bf16, layout (B,T,H,D)) ----------
__global__ void qbias_kernel(const bf16* __restrict__ qkv, const float* __restrict__ u,
                             const float* __restrict__ v, bf16* __restrict__ qu,
                             bf16* __restrict__ qv) {
    int idx = blockIdx.x * 256 + threadIdx.x;      // B*T*H*D = 2^21
    if (idx >= B_ * T_ * H_ * D_) return;
    int d = idx & 63, hh = (idx >> 6) & 15, i = (idx >> 10) & 511, b = idx >> 19;
    float qf = bf2f(qkv[((size_t)(b * L_ + TAU_ + i)) * 3072 + hh * 64 + d]);
    qu[idx] = f2bf(qf + u[hh * 64 + d]);
    qv[idx] = f2bf(qf + v[hh * 64 + d]);
}

// ---------- V transpose: vt[b][h][d][j] ----------
__global__ void vt_kernel(const bf16* __restrict__ qkv, bf16* __restrict__ vt) {
    int idx = blockIdx.x * 256 + threadIdx.x;      // B*H*D*L = 2^22
    if (idx >= B_ * H_ * D_ * L_) return;
    int j = idx & 1023, d = (idx >> 10) & 63, hh = (idx >> 16) & 15, b = idx >> 20;
    vt[idx] = qkv[((size_t)(b * L_ + j)) * 3072 + 2048 + hh * 64 + d];
}

// ---------- fused flash attention with relative shift ----------
__global__ void __launch_bounds__(32)
attn_kernel(const bf16* __restrict__ qu, const bf16* __restrict__ qv,
            const bf16* __restrict__ qkv, const bf16* __restrict__ vt,
            const bf16* __restrict__ Rm, bf16* __restrict__ attn) {
    __shared__ __align__(16) unsigned short plds[256];
    int lane = threadIdx.x;
    int i0 = blockIdx.x * 16, h = blockIdx.y, b = blockIdx.z;
    int m15 = lane & 15, half = lane >> 4;
    int kb_a = half * 16, kb_b = half * 16;

    // A-operands for (q+u) and (q+v): 16x64 tiles, d split into two K=32 chunks
    v16bf a_qu[2], a_qv[2];
#pragma unroll
    for (int t = 0; t < 2; ++t) {
        size_t base = (((size_t)(b * T_ + i0 + m15)) * H_ + h) * 64 + t * 32 + kb_a;
        v8bf lo = *(const v8bf*)(qu + base), hi = *(const v8bf*)(qu + base + 8);
#pragma unroll
        for (int i = 0; i < 8; ++i) { a_qu[t][i] = lo[i]; a_qu[t][i + 8] = hi[i]; }
        v8bf lo2 = *(const v8bf*)(qv + base), hi2 = *(const v8bf*)(qv + base + 8);
#pragma unroll
        for (int i = 0; i < 8; ++i) { a_qv[t][i] = lo2[i]; a_qv[t][i + 8] = hi2[i]; }
    }

    v8f o0 = {}, o1 = {}, o2 = {}, o3 = {};
    float mrow[8], lrow[8];
#pragma unroll
    for (int r = 0; r < 8; ++r) { mrow[r] = -3.0e38f; lrow[r] = 0.f; }

    const bf16 bz = __builtin_bit_cast(bf16, (unsigned short)0);
    int jhi = TAU_ + i0 + 15;
    int njt = jhi / 16 + 1;
    for (int jt = 0; jt < njt; ++jt) {
        int j0 = jt * 16;
        // ---- score tile S = (q+u) K^T ----
        v8f cs = {};
#pragma unroll
        for (int t = 0; t < 2; ++t) {
            const bf16* kp = qkv + ((size_t)(b * L_ + j0 + m15)) * 3072 + 1024 + h * 64 + t * 32 + kb_b;
            cs = wmma_bf16(a_qu[t], *(const v16bf*)kp, cs);
        }
        // ---- pos tiles: columns jj = csc..csc+31 ----
        int csc = j0 - i0 + 496;
        v8f cp0 = {}, cp1 = {};
        int c0 = csc + m15;      if (c0 > 1023) c0 = 1023;
        int c1 = csc + 16 + m15; if (c1 > 1023) c1 = 1023;
#pragma unroll
        for (int t = 0; t < 2; ++t) {
            const bf16* r0 = Rm + (size_t)c0 * DM_ + h * 64 + t * 32 + kb_b;
            const bf16* r1 = Rm + (size_t)c1 * DM_ + h * 64 + t * 32 + kb_b;
            cp0 = wmma_bf16(a_qv[t], *(const v16bf*)r0, cp0);
            cp1 = wmma_bf16(a_qv[t], *(const v16bf*)r1, cp1);
        }
        // ---- rel-shift extract + mask + online softmax ----
        float p[8];
#pragma unroll
        for (int r = 0; r < 8; ++r) {
            int c = m15 + (half ? (7 - r) : (15 - r));   // in [0,30]
            int srcl = (c & 15) | (lane & 16);
            float v0 = bperm(cp0[r], srcl);
            float v1 = bperm(cp1[r], srcl);
            float pos = (c < 16) ? v0 : v1;
            float sval = (cs[r] + pos) * 0.125f;         // 1/sqrt(64)
            int irow = i0 + r + half * 8;
            int jcol = j0 + m15;
            if (jcol > TAU_ + irow) sval = -1e30f;
            float mx = sval;
            mx = fmaxf(mx, bperm(mx, lane ^ 1));
            mx = fmaxf(mx, bperm(mx, lane ^ 2));
            mx = fmaxf(mx, bperm(mx, lane ^ 4));
            mx = fmaxf(mx, bperm(mx, lane ^ 8));
            float mnew = fmaxf(mrow[r], mx);
            float alpha = __expf(mrow[r] - mnew);
            float pv = __expf(sval - mnew);
            float rs = pv;
            rs += bperm(rs, lane ^ 1);
            rs += bperm(rs, lane ^ 2);
            rs += bperm(rs, lane ^ 4);
            rs += bperm(rs, lane ^ 8);
            lrow[r] = lrow[r] * alpha + rs;
            mrow[r] = mnew;
            o0[r] *= alpha; o1[r] *= alpha; o2[r] *= alpha; o3[r] *= alpha;
            p[r] = pv;
        }
        // ---- P (C-layout) -> LDS -> A-layout (K padded to 32 with zeros) ----
#pragma unroll
        for (int r = 0; r < 8; ++r)
            plds[(r + half * 8) * 16 + m15] = __builtin_bit_cast(unsigned short, f2bf(p[r]));
        __syncthreads();
        v16bf pa;
        {
            v8bf pl = *(const v8bf*)&plds[m15 * 16];
            v8bf ph = *(const v8bf*)&plds[m15 * 16 + 8];
#pragma unroll
            for (int i = 0; i < 8; ++i) { pa[i] = half ? bz : pl[i]; pa[i + 8] = half ? bz : ph[i]; }
        }
        __syncthreads();
        // ---- O += P * V  (V^T precomputed: vt[b][h][d][j]) ----
        int jr = j0 + kb_b; if (jr > L_ - 16) jr = L_ - 16;
        const bf16* vb = vt + ((size_t)(b * H_ + h)) * 64 * L_;
        o0 = wmma_bf16(pa, *(const v16bf*)(vb + (size_t)(0 * 16 + m15) * L_ + jr), o0);
        o1 = wmma_bf16(pa, *(const v16bf*)(vb + (size_t)(1 * 16 + m15) * L_ + jr), o1);
        o2 = wmma_bf16(pa, *(const v16bf*)(vb + (size_t)(2 * 16 + m15) * L_ + jr), o2);
        o3 = wmma_bf16(pa, *(const v16bf*)(vb + (size_t)(3 * 16 + m15) * L_ + jr), o3);
    }
    // ---- epilogue: normalize, write attn (B,T,H*D) bf16 ----
#pragma unroll
    for (int r = 0; r < 8; ++r) {
        float inv = 1.0f / lrow[r];
        int irow = i0 + r + half * 8;
        bf16* op = attn + ((size_t)(b * T_ + irow)) * (H_ * D_) + h * 64;
        op[0 * 16 + m15] = f2bf(o0[r] * inv);
        op[1 * 16 + m15] = f2bf(o1[r] * inv);
        op[2 * 16 + m15] = f2bf(o2[r] * inv);
        op[3 * 16 + m15] = f2bf(o3[r] * inv);
    }
}

// ---------- host launcher ----------
extern "C" void kernel_launch(void* const* d_in, const int* in_sizes, int n_in,
                              void* d_out, int out_size, void* d_ws, size_t ws_size,
                              hipStream_t stream) {
    const float* inputs = (const float*)d_in[0];
    const float* memory = (const float*)d_in[1];
    const float* w_qkv  = (const float*)d_in[2];
    const float* w_pos  = (const float*)d_in[3];
    const float* w_out  = (const float*)d_in[4];
    const float* uvar   = (const float*)d_in[5];
    const float* vvar   = (const float*)d_in[6];
    const float* gamma  = (const float*)d_in[7];
    const float* beta   = (const float*)d_in[8];

    char* ws = (char*)d_ws;
    size_t off = 0;
    auto take = [&](size_t bytes) { char* p = ws + off; off += (bytes + 255) & ~(size_t)255; return p; };

    bf16* xln   = (bf16*)take((size_t)B_ * L_ * DM_ * 2);        // 8 MB
    bf16* phi   = (bf16*)take((size_t)L_ * DM_ * 2);             // 2 MB
    bf16* wq_b  = (bf16*)take((size_t)3 * DM_ * DM_ * 2);        // 6 MB
    bf16* wp_b  = (bf16*)take((size_t)DM_ * DM_ * 2);            // 2 MB
    bf16* wo_b  = (bf16*)take((size_t)DM_ * DM_ * 2);            // 2 MB
    bf16* qkv   = (bf16*)take((size_t)B_ * L_ * 3 * DM_ * 2);    // 24 MB
    bf16* qu    = (bf16*)take((size_t)B_ * T_ * H_ * D_ * 2);    // 4 MB
    bf16* qv    = (bf16*)take((size_t)B_ * T_ * H_ * D_ * 2);    // 4 MB
    bf16* vt    = (bf16*)take((size_t)B_ * H_ * D_ * L_ * 2);    // 8 MB
    bf16* Rm    = (bf16*)take((size_t)L_ * DM_ * 2);             // 2 MB
    bf16* attn  = (bf16*)take((size_t)B_ * T_ * H_ * D_ * 2);    // 4 MB

    // 1. LayerNorm (+bf16)
    ln_kernel<<<B_ * L_, 256, 0, stream>>>(inputs, memory, gamma, beta, xln);
    // 2. positional basis
    phi_kernel<<<L_, 256, 0, stream>>>(phi);
    // 3. weights -> bf16
    cvt_kernel<<<(3 * DM_ * DM_ + 255) / 256, 256, 0, stream>>>(w_qkv, wq_b, 3 * DM_ * DM_);
    cvt_kernel<<<(DM_ * DM_ + 255) / 256, 256, 0, stream>>>(w_pos, wp_b, DM_ * DM_);
    cvt_kernel<<<(DM_ * DM_ + 255) / 256, 256, 0, stream>>>(w_out, wo_b, DM_ * DM_);
    // 4. qkv = xln @ w_qkv^T   (M=4096, N=3072, K=1024)
    gemm_nt<<<dim3(3 * DM_ / 64, B_ * L_ / 64), 128, 0, stream>>>(
        xln, wq_b, qkv, B_ * L_, 3 * DM_, DM_, DM_, DM_, 3 * DM_, 0);
    // 5. R = phi @ w_pos^T     (M=1024, N=1024, K=1024)
    gemm_nt<<<dim3(DM_ / 64, L_ / 64), 128, 0, stream>>>(
        phi, wp_b, Rm, L_, DM_, DM_, DM_, DM_, DM_, 0);
    // 6. q + u / q + v
    qbias_kernel<<<(B_ * T_ * H_ * D_) / 256, 256, 0, stream>>>(qkv, uvar, vvar, qu, qv);
    // 7. V^T
    vt_kernel<<<(B_ * H_ * D_ * L_) / 256, 256, 0, stream>>>(qkv, vt);
    // 8. fused attention (flash-style, rel-shift in-register)
    attn_kernel<<<dim3(T_ / 16, H_, B_), 32, 0, stream>>>(qu, qv, qkv, vt, Rm, attn);
    // 9. final projection -> f32 d_out (M=2048, N=1024, K=1024)
    gemm_nt<<<dim3(DM_ / 64, B_ * T_ / 64), 128, 0, stream>>>(
        attn, wo_b, d_out, B_ * T_, DM_, DM_, DM_, DM_, DM_, 1);
    (void)n_in; (void)in_sizes; (void)out_size; (void)ws_size;
}
